// SelfAttention_SD_77275051590041
// MI455X (gfx1250) — matrix operand order repaired and checked by reading, not compiled, not run on hardware
//
#include <hip/hip_runtime.h>

typedef __attribute__((ext_vector_type(16))) __bf16 v16bf;
typedef __attribute__((ext_vector_type(8)))  __bf16 v8bf;
typedef __attribute__((ext_vector_type(8)))  float  v8f;

#define DIMX   1024
#define HEADS  16
#define DH     64
#define NSEQ   2048
#define BATCH  4

__device__ __forceinline__ v8f wmma_bf16(v16bf a, v16bf b, v8f c) {
  return __builtin_amdgcn_wmma_f32_16x16x32_bf16(false, a, false, b, (short)0, c, false, false);
}

// ---- WMMA fragment loaders (wave32 layouts per CDNA5 ISA 7.12.2) ----
// A fragment 16x32 bf16: lane L -> row L%16; elems e<8 -> k = (L/16)*8 + e, e>=8 -> k+16.
__device__ __forceinline__ v16bf load_a_bf16(const __bf16* p, int ld, int lane) {
  int hf = lane >> 4, r = lane & 15;
  const __bf16* rp = p + (size_t)r * ld + hf * 8;
  v16bf a;
#pragma unroll
  for (int e = 0; e < 8; ++e) { a[e] = rp[e]; a[8 + e] = rp[16 + e]; }
  return a;
}
__device__ __forceinline__ v16bf load_a_f32(const float* p, int ld, int lane) {
  int hf = lane >> 4, r = lane & 15;
  const float* rp = p + (size_t)r * ld + hf * 8;
  v16bf a;
#pragma unroll
  for (int e = 0; e < 8; ++e) { a[e] = (__bf16)rp[e]; a[8 + e] = (__bf16)rp[16 + e]; }
  return a;
}
// B fragment 32x16 bf16 where B(k,n) = p[(n0+n)*ld + k0+k]: lane L -> col L%16; elems e<8 -> k = (L/16)*8 + e, e>=8 -> k+16.
__device__ __forceinline__ v16bf load_b_bf16(const __bf16* p, int ld, int lane) {
  int hf = lane >> 4, n = lane & 15;
  const __bf16* rp = p + (size_t)n * ld + hf * 8;
  v16bf b;
#pragma unroll
  for (int e = 0; e < 8; ++e) { b[e] = rp[e]; b[8 + e] = rp[16 + e]; }
  return b;
}
__device__ __forceinline__ v16bf load_b_f32(const float* p, int ld, int lane) {
  int hf = lane >> 4, n = lane & 15;
  const float* rp = p + (size_t)n * ld + hf * 8;
  v16bf b;
#pragma unroll
  for (int e = 0; e < 8; ++e) { b[e] = (__bf16)rp[e]; b[8 + e] = (__bf16)rp[16 + e]; }
  return b;
}

// ---- Kernel 1: fused QKV projection, fp32 in -> bf16 out (Q,K head-major; V transposed) ----
// grid.x = 3072/64 col tiles, grid.y = 8192/128 row tiles, 256 threads = 8 waves (4M x 2N), 32x32/wave
// Q/K tiles computed "transposed" (A = weight rows, B = x^T) so each lane's 8 accum
// elements are 8 consecutive d at fixed seq -> one 16B store into [B,H,N,D].
// V tiles computed normally (A = x, B = W^T) so the 8 elements are 8 consecutive seq
// at fixed d -> one 16B store into transposed Vt[B,H,D,N].
__global__ __launch_bounds__(256) void qkv_kernel(
    const float* __restrict__ x, const float* __restrict__ Wq,
    const float* __restrict__ Wk, const float* __restrict__ Wv,
    __bf16* __restrict__ qws, __bf16* __restrict__ kws, __bf16* __restrict__ vtws) {
  int lane = threadIdx.x & 31, wave = threadIdx.x >> 5;
  int hf = lane >> 4, lr = lane & 15;
  int row0 = blockIdx.y * 128 + (wave >> 1) * 32;
  int col0 = blockIdx.x * 64 + (wave & 1) * 32;
  int which = col0 >> 10;                     // 0=Q 1=K 2=V (uniform per block)
  int wrow0 = col0 & 1023;

  v8f c00 = {}, c01 = {}, c10 = {}, c11 = {};
  if (which < 2) {
    const float* W = (which == 0) ? Wq : Wk;
    __bf16* dst = (which == 0) ? qws : kws;
    for (int k0 = 0; k0 < DIMX; k0 += 32) {
      if (k0 + 32 < DIMX) {
        __builtin_prefetch(W + (size_t)wrow0 * DIMX + k0 + 32, 0, 3);
        __builtin_prefetch(x + (size_t)row0 * DIMX + k0 + 32, 0, 3);
      }
      v16bf aw0 = load_a_f32(W + (size_t)wrow0 * DIMX + k0, DIMX, lane);
      v16bf aw1 = load_a_f32(W + (size_t)(wrow0 + 16) * DIMX + k0, DIMX, lane);
      v16bf bx0 = load_b_f32(x + (size_t)row0 * DIMX + k0, DIMX, lane);
      v16bf bx1 = load_b_f32(x + (size_t)(row0 + 16) * DIMX + k0, DIMX, lane);
      c00 = wmma_bf16(aw0, bx0, c00);
      c01 = wmma_bf16(aw0, bx1, c01);
      c10 = wmma_bf16(aw1, bx0, c10);
      c11 = wmma_bf16(aw1, bx1, c11);
    }
    v8f acc[2][2] = {{c00, c01}, {c10, c11}};
#pragma unroll
    for (int mi = 0; mi < 2; ++mi)       // output-channel tile
#pragma unroll
      for (int ni = 0; ni < 2; ++ni) {   // seq tile
        int m = row0 + ni * 16 + lr;     // global row = b*2048 + seq (lane-varying)
        int bb = m >> 11, seq = m & 2047;
        int cgb = wrow0 + mi * 16 + 8 * hf;      // 8-aligned channel base
        int head = cgb >> 6, db = cgb & 63;
        v8bf pk;
#pragma unroll
        for (int i = 0; i < 8; ++i) pk[i] = (__bf16)acc[mi][ni][i];
        *(v8bf*)&dst[((size_t)(bb * HEADS + head) * NSEQ + seq) * DH + db] = pk;
      }
  } else {
    for (int k0 = 0; k0 < DIMX; k0 += 32) {
      if (k0 + 32 < DIMX) {
        __builtin_prefetch(Wv + (size_t)wrow0 * DIMX + k0 + 32, 0, 3);
        __builtin_prefetch(x + (size_t)row0 * DIMX + k0 + 32, 0, 3);
      }
      v16bf ax0 = load_a_f32(x + (size_t)row0 * DIMX + k0, DIMX, lane);
      v16bf ax1 = load_a_f32(x + (size_t)(row0 + 16) * DIMX + k0, DIMX, lane);
      v16bf bw0 = load_b_f32(Wv + (size_t)wrow0 * DIMX + k0, DIMX, lane);
      v16bf bw1 = load_b_f32(Wv + (size_t)(wrow0 + 16) * DIMX + k0, DIMX, lane);
      c00 = wmma_bf16(ax0, bw0, c00);
      c01 = wmma_bf16(ax0, bw1, c01);
      c10 = wmma_bf16(ax1, bw0, c10);
      c11 = wmma_bf16(ax1, bw1, c11);
    }
    v8f acc[2][2] = {{c00, c01}, {c10, c11}};
#pragma unroll
    for (int mi = 0; mi < 2; ++mi)       // seq tile
#pragma unroll
      for (int ni = 0; ni < 2; ++ni) {   // output-channel tile
        int cg = wrow0 + ni * 16 + lr;   // channel (lane-varying)
        int head = cg >> 6, d = cg & 63;
        int mb = row0 + mi * 16 + 8 * hf;        // 8-aligned seq base
        int bb = mb >> 11, seqb = mb & 2047;
        v8bf pk;
#pragma unroll
        for (int i = 0; i < 8; ++i) pk[i] = (__bf16)acc[mi][ni][i];
        *(v8bf*)&vtws[((size_t)(bb * HEADS + head) * DH + d) * NSEQ + seqb] = pk;
      }
  }
}

// ---- Kernel 2: flash attention per (b,h), 4 waves x 16 query rows, 64-key blocks ----
// grid.x = B*H, grid.y = N/64, 128 threads
__global__ __launch_bounds__(128) void attn_kernel(
    const __bf16* __restrict__ qws, const __bf16* __restrict__ kws,
    const __bf16* __restrict__ vtws, __bf16* __restrict__ ows) {
  int bh = blockIdx.x;
  int b = bh >> 4, h = bh & 15;
  int wave = threadIdx.x >> 5, lane = threadIdx.x & 31;
  int hf = lane >> 4, lr = lane & 15;
  int q0 = blockIdx.y * 64 + wave * 16;

  const __bf16* Qh  = qws  + (size_t)bh * NSEQ * DH;
  const __bf16* Kh  = kws  + (size_t)bh * NSEQ * DH;
  const __bf16* Vth = vtws + (size_t)bh * DH * NSEQ;

  __shared__ float  sS[4][16][66];     // padded vs bank conflicts
  __shared__ __bf16 sP[4][16][64];
  __shared__ float  sAlpha[4][16];
  __shared__ float  sL[4][16];

  v16bf aq0 = load_a_bf16(Qh + (size_t)q0 * DH + 0,  DH, lane);
  v16bf aq1 = load_a_bf16(Qh + (size_t)q0 * DH + 32, DH, lane);

  v8f o[4] = {v8f{}, v8f{}, v8f{}, v8f{}};
  float m_run = -3.0e38f, l_run = 0.0f;   // valid in lanes 0..15 (row = lane)
  const float scale = 0.125f;             // 64^-0.5

  for (int key0 = 0; key0 < NSEQ; key0 += 64) {
    // S tile = Q @ K^T (16 x 64), fp32
#pragma unroll
    for (int j = 0; j < 4; ++j) {
      v16bf bk0 = load_b_bf16(Kh + (size_t)(key0 + j * 16) * DH + 0,  DH, lane);
      v16bf bk1 = load_b_bf16(Kh + (size_t)(key0 + j * 16) * DH + 32, DH, lane);
      v8f s = {};
      s = wmma_bf16(aq0, bk0, s);
      s = wmma_bf16(aq1, bk1, s);
#pragma unroll
      for (int i = 0; i < 8; ++i) sS[wave][i + 8 * hf][j * 16 + lr] = s[i] * scale;
    }
    __syncthreads();
    // online softmax, lanes 0..15 each own one row
    if (lane < 16) {
      int r = lane;
      float mb = -3.0e38f;
      for (int c = 0; c < 64; ++c) mb = fmaxf(mb, sS[wave][r][c]);
      float mn = fmaxf(m_run, mb);
      float alpha = __expf(m_run - mn);
      float sum = 0.0f;
      for (int c = 0; c < 64; ++c) {
        float p = __expf(sS[wave][r][c] - mn);
        sum += p;
        sP[wave][r][c] = (__bf16)p;
      }
      l_run = l_run * alpha + sum;
      m_run = mn;
      sAlpha[wave][r] = alpha;
    }
    __syncthreads();
    // rescale O accumulators, then O += P @ V
#pragma unroll
    for (int i = 0; i < 8; ++i) {
      float al = sAlpha[wave][i + 8 * hf];
      o[0][i] *= al; o[1][i] *= al; o[2][i] *= al; o[3][i] *= al;
    }
    v16bf ap0 = load_a_bf16(&sP[wave][0][0] + 0,  64, lane);
    v16bf ap1 = load_a_bf16(&sP[wave][0][0] + 32, 64, lane);
#pragma unroll
    for (int j = 0; j < 4; ++j) {
      v16bf bv0 = load_b_bf16(Vth + (size_t)(j * 16) * NSEQ + key0 + 0,  NSEQ, lane);
      v16bf bv1 = load_b_bf16(Vth + (size_t)(j * 16) * NSEQ + key0 + 32, NSEQ, lane);
      o[j] = wmma_bf16(ap0, bv0, o[j]);
      o[j] = wmma_bf16(ap1, bv1, o[j]);
    }
  }
  if (lane < 16) sL[wave][lane] = l_run;
  __syncthreads();
#pragma unroll
  for (int j = 0; j < 4; ++j)
#pragma unroll
    for (int i = 0; i < 8; ++i) {
      int r = i + 8 * hf;
      float denom = sL[wave][r] + 1e-10f;
      int seq = q0 + r;
      ows[((size_t)(b * NSEQ + seq)) * DIMX + h * DH + j * 16 + lr] = (__bf16)(o[j][i] / denom);
    }
}

// ---- Kernel 3: output projection O @ Wo^T + bo, fp32 out ----
// grid.x = 1024/64, grid.y = 8192/128, 256 threads = 8 waves (4M x 2N), 32x32/wave
// Transposed orientation (A = Wo rows, B = O^T): each lane's 8 accum elements are
// 8 consecutive output channels at fixed row -> one 32B vector store + vector bias load.
__global__ __launch_bounds__(256) void proj_kernel(
    const __bf16* __restrict__ O, const float* __restrict__ Wo,
    const float* __restrict__ bo, float* __restrict__ out) {
  int lane = threadIdx.x & 31, wave = threadIdx.x >> 5;
  int hf = lane >> 4, lr = lane & 15;
  int row0 = blockIdx.y * 128 + (wave >> 1) * 32;
  int col0 = blockIdx.x * 64 + (wave & 1) * 32;

  v8f c00 = {}, c01 = {}, c10 = {}, c11 = {};
  for (int k0 = 0; k0 < DIMX; k0 += 32) {
    if (k0 + 32 < DIMX) {
      __builtin_prefetch(Wo + (size_t)col0 * DIMX + k0 + 32, 0, 3);
      __builtin_prefetch(O + (size_t)row0 * DIMX + k0 + 32, 0, 3);
    }
    v16bf aw0 = load_a_f32(Wo + (size_t)col0 * DIMX + k0, DIMX, lane);
    v16bf aw1 = load_a_f32(Wo + (size_t)(col0 + 16) * DIMX + k0, DIMX, lane);
    v16bf bo0 = load_b_bf16(O + (size_t)row0 * DIMX + k0, DIMX, lane);
    v16bf bo1 = load_b_bf16(O + (size_t)(row0 + 16) * DIMX + k0, DIMX, lane);
    c00 = wmma_bf16(aw0, bo0, c00);
    c01 = wmma_bf16(aw0, bo1, c01);
    c10 = wmma_bf16(aw1, bo0, c10);
    c11 = wmma_bf16(aw1, bo1, c11);
  }
  v8f acc[2][2] = {{c00, c01}, {c10, c11}};
#pragma unroll
  for (int mi = 0; mi < 2; ++mi) {       // output-channel tile
    int cgb = col0 + mi * 16 + 8 * hf;   // 8-aligned channel base
    v8f bias = *(const v8f*)&bo[cgb];
#pragma unroll
    for (int ni = 0; ni < 2; ++ni) {     // row tile
      int m = row0 + ni * 16 + lr;       // lane-varying row
      v8f r;
#pragma unroll
      for (int i = 0; i < 8; ++i) r[i] = acc[mi][ni][i] + bias[i];
      *(v8f*)&out[(size_t)m * DIMX + cgb] = r;
    }
  }
}

extern "C" void kernel_launch(void* const* d_in, const int* in_sizes, int n_in,
                              void* d_out, int out_size, void* d_ws, size_t ws_size,
                              hipStream_t stream) {
  (void)in_sizes; (void)n_in; (void)out_size; (void)ws_size;
  const float* x  = (const float*)d_in[0];
  const float* Wq = (const float*)d_in[1];
  const float* Wk = (const float*)d_in[2];
  const float* Wv = (const float*)d_in[3];
  const float* Wo = (const float*)d_in[4];
  const float* bo = (const float*)d_in[5];
  float* out = (float*)d_out;

  const size_t QKV_ELEMS = (size_t)BATCH * HEADS * NSEQ * DH;  // 8 Mi elems (16 MB bf16)
  __bf16* qws  = (__bf16*)d_ws;
  __bf16* kws  = qws + QKV_ELEMS;
  __bf16* vtws = kws + QKV_ELEMS;
  __bf16* ows  = vtws + QKV_ELEMS;

  // 1) QKV projections: [8192,1024] x [1024,3072]
  qkv_kernel<<<dim3(48, 64), 256, 0, stream>>>(x, Wq, Wk, Wv, qws, kws, vtws);
  // 2) flash attention per (b,h)
  attn_kernel<<<dim3(BATCH * HEADS, NSEQ / 64), 128, 0, stream>>>(qws, kws, vtws, ows);
  // 3) output projection + bias
  proj_kernel<<<dim3(16, 64), 256, 0, stream>>>(ows, Wo, bo, out);
}
